// DeepSeekV3_1Attention_2001454760761
// MI455X (gfx1250) — matrix operand order repaired and checked by reading, not compiled, not run on hardware
//
#include <hip/hip_runtime.h>

// ---------------- problem constants ----------------
#define B_   2
#define S_   2048
#define D_   2048
#define H_   16
#define NOPE_ 128
#define ROPE_ 64
#define VD_  128
#define QL_  512
#define KVL_ 512
#define HR_  (H_ * ROPE_)    // 1024
#define HN_  (H_ * NOPE_)    // 2048
#define HV_  (H_ * VD_)      // 2048
#define NTOK_ (B_ * S_)      // 4096
#define SCALE_ 0.07216878364870322f   // 1/sqrt(192)

typedef __attribute__((ext_vector_type(16))) __bf16          v16bf;
typedef __attribute__((ext_vector_type(16))) unsigned short  v16us;
typedef __attribute__((ext_vector_type(8)))  float           v8f;

// ---------------- helpers ----------------
__device__ __forceinline__ unsigned short f2bf(float x) {
  unsigned int u = __float_as_uint(x);
  unsigned int r = u + 0x7FFFu + ((u >> 16) & 1u);
  return (unsigned short)(r >> 16);
}

__device__ __forceinline__ v8f wmma_bf16(v16us a, v16us b, v8f c) {
  return __builtin_amdgcn_wmma_f32_16x16x32_bf16(
      false, __builtin_bit_cast(v16bf, a),
      false, __builtin_bit_cast(v16bf, b),
      (short)0, c, false, false);
}

// A fragment: 16x32 bf16, row-major source with leading dim ld (elements).
// lane<16: row=lane, K=0..7 then 16..23 ; lane>=16: row=lane-16, K=8..15 then 24..31
__device__ __forceinline__ v16us load_frag_A(const unsigned short* __restrict__ base, int ld) {
  int lane = threadIdx.x & 31;
  int r  = lane & 15;
  int kb = (lane >> 4) * 8;
  const unsigned short* p = base + (long)r * ld + kb;
  v16us f;
#pragma unroll
  for (int i = 0; i < 8; ++i) f[i] = p[i];
#pragma unroll
  for (int i = 0; i < 8; ++i) f[8 + i] = p[16 + i];
  return f;
}

// B fragment from "Bt" layout: Bt is N x K row-major (so B[k][n] = Bt[n*ld + k]).
// lane<16: col n=lane, K=0..7 then 16..23 ; lane>=16: col n=lane-16, K=8..15 then 24..31
__device__ __forceinline__ v16us load_frag_Bt(const unsigned short* __restrict__ base, int ld) {
  int lane = threadIdx.x & 31;
  int n  = lane & 15;
  int kb = (lane >> 4) * 8;
  const unsigned short* p = base + (long)n * ld + kb;
  v16us f;
#pragma unroll
  for (int i = 0; i < 8; ++i) f[i] = p[i];
#pragma unroll
  for (int i = 0; i < 8; ++i) f[8 + i] = p[16 + i];
  return f;
}

__device__ __forceinline__ float halfmax(float v) {
#pragma unroll
  for (int o = 8; o >= 1; o >>= 1) v = fmaxf(v, __shfl_xor(v, o, 32));
  return v;
}
__device__ __forceinline__ float halfsum(float v) {
#pragma unroll
  for (int o = 8; o >= 1; o >>= 1) v += __shfl_xor(v, o, 32);
  return v;
}

// ---------------- conversion / transpose kernels ----------------
__global__ void k_convert_bf16(const float* __restrict__ in, unsigned short* __restrict__ out, long n) {
  long i = (long)blockIdx.x * blockDim.x + threadIdx.x;
  if (i < n) out[i] = f2bf(in[i]);
}

// in: z-batched (R x C) f32 row-major -> out: (C x R) bf16 row-major
__global__ void k_transpose_bf16(const float* __restrict__ in, unsigned short* __restrict__ out,
                                 int R, int C, long inZ, long outZ) {
  int z = blockIdx.z;
  long i = (long)blockIdx.x * blockDim.x + threadIdx.x;
  long total = (long)R * C;
  if (i >= total) return;
  int r = (int)(i / C), c = (int)(i % C);
  out[(long)z * outZ + (long)c * R + r] = f2bf(in[(long)z * inZ + i]);
}

// bf16 (R x C) -> bf16 (C x R), z-batched
__global__ void k_transpose_us(const unsigned short* __restrict__ in, unsigned short* __restrict__ out,
                               int R, int C, long inZ, long outZ) {
  int z = blockIdx.z;
  long i = (long)blockIdx.x * blockDim.x + threadIdx.x;
  long total = (long)R * C;
  if (i >= total) return;
  int r = (int)(i / C), c = (int)(i % C);
  out[(long)z * outZ + (long)c * R + r] = in[(long)z * inZ + i];
}

// ---------------- YaRN RoPE ----------------
// in/out: (nTok, nHead, 64); position = tok % S_
__global__ void k_rope(const float* __restrict__ in, unsigned short* __restrict__ out,
                       int nTok, int nHead) {
  long idx = (long)blockIdx.x * blockDim.x + threadIdx.x;
  long total = (long)nTok * nHead * 32;
  if (idx >= total) return;
  int i   = (int)(idx % 32);
  long r  = idx / 32;
  int hh  = (int)(r % nHead);
  int tok = (int)(r / nHead);
  int pos = tok % S_;
  float freq = powf(10000.0f, -(float)(2 * i) / 64.0f);
  float wl   = 2.0f * 3.14159265358979f / freq;
  float ramp = (wl / 4096.0f - 1.0f) / 31.0f;
  ramp = fminf(fmaxf(ramp, 0.0f), 1.0f);
  float sc   = 1.0f - ramp + ramp * 40.0f;
  float fr   = (float)pos * (freq / sc);
  float cv = cosf(fr), sv = sinf(fr);
  long base = ((long)tok * nHead + hh) * 64;
  float x1 = in[base + i], x2 = in[base + 32 + i];
  out[base + i]      = f2bf(x1 * cv - x2 * sv);
  out[base + 32 + i] = f2bf(x1 * sv + x2 * cv);
}

// ---------------- register-blocked WMMA GEMM ----------------
// Each wave computes a 64x64 macro-tile (4x4 wmma tiles); 4 waves/block -> 128x128 block tile.
// C[m,n] = sum_k A[m,k]*Bt[n,k] + bias[n]; z-batched with (b,h) = (z/Hdim, z%Hdim)
__global__ __launch_bounds__(128) void k_gemm_bf16(
    const unsigned short* __restrict__ A, const unsigned short* __restrict__ Bt,
    const float* __restrict__ bias,
    float* __restrict__ Cf, unsigned short* __restrict__ Cb,
    int M, int N, int K, int lda, int ldb, int ldc,
    long aZb, long aZh, long bZh, long cZb, long cZh, int Hdim) {
  int wave = threadIdx.x >> 5;
  int lane = threadIdx.x & 31;
  int m0 = blockIdx.y * 128 + (wave >> 1) * 64;
  int n0 = blockIdx.x * 128 + (wave & 1) * 64;
  if (m0 >= M || n0 >= N) return;
  int z  = blockIdx.z;
  int zb = z / Hdim, zh = z % Hdim;
  const unsigned short* Ap = A + (long)zb * aZb + (long)zh * aZh + (long)m0 * lda;
  const unsigned short* Bp = Bt + (long)zh * bZh + (long)n0 * ldb;

  v8f acc[4][4];
#pragma unroll
  for (int i = 0; i < 4; ++i)
#pragma unroll
    for (int j = 0; j < 4; ++j) acc[i][j] = (v8f){0.f,0.f,0.f,0.f,0.f,0.f,0.f,0.f};

  for (int k = 0; k < K; k += 32) {
    v16us af[4], bf[4];
#pragma unroll
    for (int i = 0; i < 4; ++i) af[i] = load_frag_A(Ap + (long)i * 16 * lda + k, lda);
#pragma unroll
    for (int j = 0; j < 4; ++j) bf[j] = load_frag_Bt(Bp + (long)j * 16 * ldb + k, ldb);
#pragma unroll
    for (int i = 0; i < 4; ++i)
#pragma unroll
      for (int j = 0; j < 4; ++j)
        acc[i][j] = wmma_bf16(af[i], bf[j], acc[i][j]);
  }

  int ln = lane & 15, hf = lane >> 4;
  long cbase = (long)zb * cZb + (long)zh * cZh;
#pragma unroll
  for (int j = 0; j < 4; ++j) {
    float bv = bias ? bias[n0 + j * 16 + ln] : 0.f;
#pragma unroll
    for (int i = 0; i < 4; ++i) {
#pragma unroll
      for (int v = 0; v < 8; ++v) {
        int row = m0 + i * 16 + v + 8 * hf;
        float x = acc[i][j][v] + bv;
        long idx = cbase + (long)row * ldc + n0 + j * 16 + ln;
        if (Cf) Cf[idx] = x;
        if (Cb) Cb[idx] = f2bf(x);
      }
    }
  }
}

// ---------------- causal flash attention ----------------
// q_pe:(B,H,S,KVL) q_rope:(B,S,H,64) keys:(B,S,KVL) k_rope:(B,S,64) v_t:(B,KVL,S) -> ctx:(B,H,S,KVL) bf16
__global__ __launch_bounds__(256) void k_attn(
    const unsigned short* __restrict__ q_pe, const unsigned short* __restrict__ q_rope,
    const unsigned short* __restrict__ keys, const unsigned short* __restrict__ k_rope,
    const unsigned short* __restrict__ v_t, unsigned short* __restrict__ ctx) {
  const int mt = blockIdx.x, h = blockIdx.y, b = blockIdx.z;
  const int s0 = mt * 16;
  const int tid = threadIdx.x;
  const int wave = tid >> 5, lane = tid & 31;
  const int ln = lane & 15, hf = lane >> 4;

  __shared__ unsigned short Pl[16 * 128];
  __shared__ float pmax[8][16];
  __shared__ float psum[8][16];
  __shared__ float rm[16], rl[16], ralpha[16];

  if (tid < 16) { rm[tid] = -3.0e38f; rl[tid] = 0.f; }
  __syncthreads();

  const unsigned short* Aq = q_pe + (((long)(b * H_ + h)) * S_ + s0) * KVL_;
  const unsigned short* Ar = q_rope + ((long)(b * S_ + s0) * H_ + h) * ROPE_;
  const unsigned short* Bk = keys + (long)b * S_ * KVL_;
  const unsigned short* Br = k_rope + (long)b * S_ * ROPE_;
  const unsigned short* Vt = v_t + (long)b * KVL_ * S_;

  // Q fragments are invariant over the T loop: cache them in registers.
  v16us qa[16], qr[2];
#pragma unroll
  for (int kk = 0; kk < 16; ++kk) qa[kk] = load_frag_A(Aq + kk * 32, KVL_);
#pragma unroll
  for (int kk = 0; kk < 2; ++kk) qr[kk] = load_frag_A(Ar + kk * 32, HR_);

  v8f accs[4];
#pragma unroll
  for (int i = 0; i < 4; ++i) accs[i] = (v8f){0.f,0.f,0.f,0.f,0.f,0.f,0.f,0.f};

  const int sLast = s0 + 15;
  for (int t0 = 0; t0 <= sLast; t0 += 128) {
    const int tw = t0 + wave * 16;
    // prefetch next T-block keys into cache (global_prefetch_b8)
    if (t0 + 128 <= sLast)
      __builtin_prefetch(Bk + (long)(tw + 128) * KVL_ + ln * 64, 0, 1);
    // ---- scores: 16x16 slice per wave (content + rope) ----
    v8f s = {0.f,0.f,0.f,0.f,0.f,0.f,0.f,0.f};
#pragma unroll
    for (int kk = 0; kk < 16; ++kk) {
      v16us bb = load_frag_Bt(Bk + (long)tw * KVL_ + kk * 32, KVL_);
      s = wmma_bf16(qa[kk], bb, s);
    }
#pragma unroll
    for (int kk = 0; kk < 2; ++kk) {
      v16us bb = load_frag_Bt(Br + (long)tw * ROPE_ + kk * 32, ROPE_);
      s = wmma_bf16(qr[kk], bb, s);
    }
    float sv[8];
#pragma unroll
    for (int v = 0; v < 8; ++v) {
      int row = s0 + v + 8 * hf;
      int col = tw + ln;
      float x = s[v] * SCALE_;
      if (col > row) x = -1.0e9f;
      sv[v] = x;
    }
    // ---- partial row max ----
#pragma unroll
    for (int v = 0; v < 8; ++v) {
      float m = halfmax(sv[v]);
      if (ln == 0) pmax[wave][v + 8 * hf] = m;
    }
    __syncthreads();                    // (a)
    if (tid < 16) {
      float mo = rm[tid], mn = mo;
#pragma unroll
      for (int w = 0; w < 8; ++w) mn = fmaxf(mn, pmax[w][tid]);
      rm[tid] = mn;
      ralpha[tid] = expf(mo - mn);
    }
    __syncthreads();                    // (b)
    // ---- p = exp(s-m), stage P in LDS, partial sums, rescale acc ----
#pragma unroll
    for (int v = 0; v < 8; ++v) {
      int r = v + 8 * hf;
      float p = expf(sv[v] - rm[r]);
      float ps = halfsum(p);
      if (ln == 0) psum[wave][r] = ps;
      Pl[r * 128 + wave * 16 + ln] = f2bf(p);
      float al = ralpha[r];
#pragma unroll
      for (int nt = 0; nt < 4; ++nt) accs[nt][v] *= al;
    }
    __syncthreads();                    // (c)
    if (tid < 16) {
      float acc = 0.f;
#pragma unroll
      for (int w = 0; w < 8; ++w) acc += psum[w][tid];
      rl[tid] = rl[tid] * ralpha[tid] + acc;
    }
    // ---- PV: each wave owns 64 output cols ----
#pragma unroll
    for (int ks = 0; ks < 4; ++ks) {
      // A fragment from LDS P tile (16 x 32 at k-offset ks*32)
      v16us a;
      {
        int r = ln, kb = hf * 8;
        const unsigned short* p = &Pl[r * 128 + ks * 32 + kb];
#pragma unroll
        for (int i = 0; i < 8; ++i) a[i] = p[i];
#pragma unroll
        for (int i = 0; i < 8; ++i) a[8 + i] = p[16 + i];
      }
#pragma unroll
      for (int nt = 0; nt < 4; ++nt) {
        int n0 = wave * 64 + nt * 16;
        v16us bb = load_frag_Bt(Vt + (long)n0 * S_ + t0 + ks * 32, S_);
        accs[nt] = wmma_bf16(a, bb, accs[nt]);
      }
    }
  }
  __syncthreads();
  // ---- epilogue: divide by row sum, write bf16 ctx ----
  float inv[8];
#pragma unroll
  for (int v = 0; v < 8; ++v) inv[v] = 1.0f / rl[v + 8 * hf];
#pragma unroll
  for (int nt = 0; nt < 4; ++nt)
#pragma unroll
    for (int v = 0; v < 8; ++v) {
      int row = s0 + v + 8 * hf;
      long idx = (((long)(b * H_ + h)) * S_ + row) * KVL_ + wave * 64 + nt * 16 + ln;
      ctx[idx] = f2bf(accs[nt][v] * inv[v]);
    }
}

// ---------------- host launch ----------------
static inline dim3 gemm_grid(int M, int N, int Z) {
  return dim3((N + 127) / 128, (M + 127) / 128, Z);
}

extern "C" void kernel_launch(void* const* d_in, const int* in_sizes, int n_in,
                              void* d_out, int out_size, void* d_ws, size_t ws_size,
                              hipStream_t stream) {
  const float* hs        = (const float*)d_in[0];
  const float* q_down_w  = (const float*)d_in[1];
  const float* q_down_b  = (const float*)d_in[2];
  const float* q_up_n_w  = (const float*)d_in[3];
  const float* q_up_n_b  = (const float*)d_in[4];
  const float* q_up_r_w  = (const float*)d_in[5];
  const float* q_up_r_b  = (const float*)d_in[6];
  const float* kv_down_w = (const float*)d_in[7];
  const float* kv_down_b = (const float*)d_in[8];
  const float* k_rope_w  = (const float*)d_in[9];
  const float* k_rope_b  = (const float*)d_in[10];
  const float* w_uk      = (const float*)d_in[11];
  const float* w_uv      = (const float*)d_in[12];
  const float* out_w     = (const float*)d_in[13];
  const float* out_b     = (const float*)d_in[14];
  float* out_f = (float*)d_out;

  size_t off = 0;
  auto alloc = [&](size_t bytes) {
    void* p = (char*)d_ws + off;
    off += (bytes + 255) & ~(size_t)255;
    return p;
  };
  typedef unsigned short us;
  us* hs_bf       = (us*)alloc((size_t)NTOK_ * D_ * 2);
  us* q_down_wt   = (us*)alloc((size_t)QL_ * D_ * 2);
  us* kv_down_wt  = (us*)alloc((size_t)KVL_ * D_ * 2);
  us* k_rope_wt   = (us*)alloc((size_t)ROPE_ * D_ * 2);
  us* q_up_n_wt   = (us*)alloc((size_t)HN_ * QL_ * 2);
  us* q_up_r_wt   = (us*)alloc((size_t)HR_ * QL_ * 2);
  us* w_uk_t      = (us*)alloc((size_t)H_ * KVL_ * NOPE_ * 2);
  us* w_uv_bf     = (us*)alloc((size_t)HV_ * KVL_ * 2);
  us* out_wt      = (us*)alloc((size_t)D_ * HV_ * 2);
  us* c_q_bf      = (us*)alloc((size_t)NTOK_ * QL_ * 2);
  us* c_kv_bf     = (us*)alloc((size_t)NTOK_ * KVL_ * 2);
  us* v_t_bf      = (us*)alloc((size_t)B_ * KVL_ * S_ * 2);
  float* k_rope_f = (float*)alloc((size_t)NTOK_ * ROPE_ * 4);
  us* k_rope_bf   = (us*)alloc((size_t)NTOK_ * ROPE_ * 2);
  float* q_rope_f = (float*)alloc((size_t)NTOK_ * HR_ * 4);
  us* q_rope_bf   = (us*)alloc((size_t)NTOK_ * HR_ * 2);
  us* q_nope_bf   = (us*)alloc((size_t)NTOK_ * HN_ * 2);
  us* q_pe_bf     = (us*)alloc((size_t)B_ * H_ * S_ * KVL_ * 2);
  us* ctx_bf      = (us*)alloc((size_t)B_ * H_ * S_ * KVL_ * 2);
  us* out_h_bf    = (us*)alloc((size_t)NTOK_ * HV_ * 2);
  (void)ws_size; (void)in_sizes; (void)n_in; (void)out_size;

  const int TB = 256;
  auto cgrid = [&](long n) { return dim3((unsigned)((n + TB - 1) / TB)); };

  // --- stage bf16 operands ---
  k_convert_bf16<<<cgrid((long)NTOK_ * D_), TB, 0, stream>>>(hs, hs_bf, (long)NTOK_ * D_);
  k_transpose_bf16<<<cgrid((long)D_ * QL_), TB, 0, stream>>>(q_down_w, q_down_wt, D_, QL_, 0, 0);
  k_transpose_bf16<<<cgrid((long)D_ * KVL_), TB, 0, stream>>>(kv_down_w, kv_down_wt, D_, KVL_, 0, 0);
  k_transpose_bf16<<<cgrid((long)D_ * ROPE_), TB, 0, stream>>>(k_rope_w, k_rope_wt, D_, ROPE_, 0, 0);
  k_transpose_bf16<<<cgrid((long)QL_ * HN_), TB, 0, stream>>>(q_up_n_w, q_up_n_wt, QL_, HN_, 0, 0);
  k_transpose_bf16<<<cgrid((long)QL_ * HR_), TB, 0, stream>>>(q_up_r_w, q_up_r_wt, QL_, HR_, 0, 0);
  {
    dim3 g = cgrid((long)NOPE_ * KVL_); g.z = H_;
    k_transpose_bf16<<<g, TB, 0, stream>>>(w_uk, w_uk_t, NOPE_, KVL_,
                                           (long)NOPE_ * KVL_, (long)KVL_ * NOPE_);
  }
  k_convert_bf16<<<cgrid((long)HV_ * KVL_), TB, 0, stream>>>(w_uv, w_uv_bf, (long)HV_ * KVL_);
  k_transpose_bf16<<<cgrid((long)HV_ * D_), TB, 0, stream>>>(out_w, out_wt, HV_, D_, 0, 0);

  // --- down projections ---
  k_gemm_bf16<<<gemm_grid(NTOK_, QL_, 1), 128, 0, stream>>>(
      hs_bf, q_down_wt, q_down_b, nullptr, c_q_bf,
      NTOK_, QL_, D_, D_, D_, QL_, 0, 0, 0, 0, 0, 1);
  k_gemm_bf16<<<gemm_grid(NTOK_, KVL_, 1), 128, 0, stream>>>(
      hs_bf, kv_down_wt, kv_down_b, nullptr, c_kv_bf,
      NTOK_, KVL_, D_, D_, D_, KVL_, 0, 0, 0, 0, 0, 1);
  k_gemm_bf16<<<gemm_grid(NTOK_, ROPE_, 1), 128, 0, stream>>>(
      hs_bf, k_rope_wt, k_rope_b, k_rope_f, nullptr,
      NTOK_, ROPE_, D_, D_, D_, ROPE_, 0, 0, 0, 0, 0, 1);

  // --- up projections ---
  k_gemm_bf16<<<gemm_grid(NTOK_, HN_, 1), 128, 0, stream>>>(
      c_q_bf, q_up_n_wt, q_up_n_b, nullptr, q_nope_bf,
      NTOK_, HN_, QL_, QL_, QL_, HN_, 0, 0, 0, 0, 0, 1);
  k_gemm_bf16<<<gemm_grid(NTOK_, HR_, 1), 128, 0, stream>>>(
      c_q_bf, q_up_r_wt, q_up_r_b, q_rope_f, nullptr,
      NTOK_, HR_, QL_, QL_, QL_, HR_, 0, 0, 0, 0, 0, 1);

  // --- RoPE ---
  k_rope<<<cgrid((long)NTOK_ * H_ * 32), TB, 0, stream>>>(q_rope_f, q_rope_bf, NTOK_, H_);
  k_rope<<<cgrid((long)NTOK_ * 32), TB, 0, stream>>>(k_rope_f, k_rope_bf, NTOK_, 1);

  // --- V^T for PV gemm ---
  {
    dim3 g = cgrid((long)S_ * KVL_); g.z = B_;
    k_transpose_us<<<g, TB, 0, stream>>>(c_kv_bf, v_t_bf, S_, KVL_,
                                         (long)S_ * KVL_, (long)KVL_ * S_);
  }

  // --- absorbed q_pe = q_nope x w_uk (per head) ---
  k_gemm_bf16<<<gemm_grid(S_, KVL_, B_ * H_), 128, 0, stream>>>(
      q_nope_bf, w_uk_t, nullptr, nullptr, q_pe_bf,
      S_, KVL_, NOPE_, HN_, NOPE_, KVL_,
      (long)S_ * HN_, (long)NOPE_, (long)KVL_ * NOPE_,
      (long)H_ * S_ * KVL_, (long)S_ * KVL_, H_);

  // --- attention ---
  k_attn<<<dim3(S_ / 16, H_, B_), 256, 0, stream>>>(
      q_pe_bf, q_rope_bf, c_kv_bf, k_rope_bf, v_t_bf, ctx_bf);

  // --- out_h = ctx x w_uv (per head) ---
  k_gemm_bf16<<<gemm_grid(S_, VD_, B_ * H_), 128, 0, stream>>>(
      ctx_bf, w_uv_bf, nullptr, nullptr, out_h_bf,
      S_, VD_, KVL_, KVL_, KVL_, HV_,
      (long)H_ * S_ * KVL_, (long)S_ * KVL_, (long)VD_ * KVL_,
      (long)S_ * HV_, (long)VD_, H_);

  // --- final projection -> f32 output ---
  k_gemm_bf16<<<gemm_grid(NTOK_, D_, 1), 128, 0, stream>>>(
      out_h_bf, out_wt, out_b, out_f, nullptr,
      NTOK_, D_, HV_, HV_, HV_, D_, 0, 0, 0, 0, 0, 1);
}
